// TopKSAE_1245540515954
// MI455X (gfx1250) — matrix operand order repaired and checked by reading, not compiled, not run on hardware
//
#include <hip/hip_runtime.h>
#include <hip/hip_bf16.h>

typedef __bf16 bf16;
typedef __attribute__((ext_vector_type(16))) __bf16 v16bf;
typedef __attribute__((ext_vector_type(8)))  __bf16 v8bf;
typedef __attribute__((ext_vector_type(8)))  float  v8f;
typedef __attribute__((ext_vector_type(4)))  unsigned int v4u;
typedef __attribute__((ext_vector_type(8)))  int v8i;
typedef __attribute__((ext_vector_type(4)))  int v4i;

#define D_MODEL 768
#define D_SAE   24576
#define NTOK    8192
#define TOPK    64

#if __has_builtin(__builtin_amdgcn_tensor_load_to_lds) && \
    __has_builtin(__builtin_amdgcn_s_wait_tensorcnt)
#define HAS_TDM 1
#endif

// ---------------------------------------------------------------------------
// Kernel 1a: convert acts f32 -> bf16 (row-major copy, one-time 25MB read)
// ---------------------------------------------------------------------------
__global__ void cvt_acts(const float* __restrict__ in, bf16* __restrict__ out) {
    int i = blockIdx.x * blockDim.x + threadIdx.x;
    if (i < NTOK * D_MODEL) out[i] = (bf16)in[i];
}

// ---------------------------------------------------------------------------
// Kernel 1b: transpose-convert W_enc [768][24576] f32 -> Wt [24576][768] bf16
// (column-major panels so WMMA B-fragments are contiguous 32B loads per lane)
// ---------------------------------------------------------------------------
__global__ void cvt_wenc(const float* __restrict__ W, bf16* __restrict__ Wt) {
    __shared__ bf16 tile[32][33];
    const int c0 = blockIdx.x * 32;
    const int k0 = blockIdx.y * 32;
    const int tx = threadIdx.x;   // 0..31
    const int ty = threadIdx.y;   // 0..7
#pragma unroll
    for (int j = 0; j < 4; ++j) {
        int k = k0 + ty + j * 8;
        tile[ty + j * 8][tx] = (bf16)W[(size_t)k * D_SAE + c0 + tx];
    }
    __syncthreads();
#pragma unroll
    for (int j = 0; j < 4; ++j) {
        int c = c0 + ty + j * 8;
        Wt[(size_t)c * D_MODEL + k0 + tx] = tile[tx][ty + j * 8];
    }
}

// ---------------------------------------------------------------------------
// wave32 min-reduction over 2 slots/lane, tracking owner (lane*2 + slot)
// ---------------------------------------------------------------------------
__device__ inline void wave_min2(float v0, float v1, int lane, float& mval, int& mid) {
    float v; int id;
    if (v1 < v0) { v = v1; id = lane * 2 + 1; } else { v = v0; id = lane * 2; }
#pragma unroll
    for (int off = 16; off; off >>= 1) {
        float ov = __shfl_xor(v, off, 32);
        int  oid = __shfl_xor(id, off, 32);
        if (ov < v) { v = ov; id = oid; }
    }
    mval = v; mid = id;
}

// ---------------------------------------------------------------------------
// Kernel 2: fused encoder GEMM (bf16 WMMA, f32 acc) + per-token top-64.
// 1 WG = 16 tokens, 8 waves. A-panel (24KB) pulled global->LDS by the Tensor
// Data Mover (TENSORcnt-tracked), then each wave: one 16x16 tile per 128-col
// step (24 x v_wmma_f32_16x16x32_bf16 along K=768), then top-k update for its
// 2 exclusively-owned tokens via LDS-staged candidates.
// Avoids materializing the 805MB pre_acts tensor entirely.
// ---------------------------------------------------------------------------
__launch_bounds__(256)
__global__ void encode_topk(const bf16* __restrict__ Abf,
                            const bf16* __restrict__ Wt,
                            const float* __restrict__ b_enc,
                            float* __restrict__ topv,
                            int*   __restrict__ topi) {
    __shared__ __align__(32) bf16  Ap[16 * D_MODEL];   // 24 KB A panel
    __shared__ __align__(16) float stage[16 * 128];     // 8 KB candidate stage

    const int tid  = threadIdx.x;
    const int lane = tid & 31;
    const int w    = tid >> 5;            // wave id 0..7
    const int hi   = (lane >> 4) & 1;     // lane half (WMMA layout)
    const int l15  = lane & 15;
    const int tok0 = blockIdx.x * 16;

#ifdef HAS_TDM
    // TDM: 1-D descriptor, 3072 x 8B = 24KB, global -> LDS. Issued by wave 0.
    if (w == 0) {
        const unsigned long long gaddr =
            (unsigned long long)(uintptr_t)(Abf + (size_t)tok0 * D_MODEL);
        const unsigned int laddr = (unsigned int)(uintptr_t)Ap;  // LDS offset
        v4u g0 = { 1u,                                   // count=1 (valid D#)
                   laddr,                                // lds_addr
                   (unsigned int)(gaddr & 0xffffffffu),
                   (unsigned int)(((gaddr >> 32) & 0x01ffffffu) | 0x80000000u) }; // type=2
        v8i g1 = { 0x00030000,          // data_size=3 (8-byte units)
                   (int)(3072u << 16),  // tensor_dim0 lo16 = 3072
                   0x00010000,          // tensor_dim0 hi16=0 | tensor_dim1 lo16=1
                   (int)(3072u << 16),  // tensor_dim1 hi16=0 | tile_dim0=3072
                   1,                   // tile_dim1=1 | tile_dim2=0
                   3072,                // tensor_dim0_stride lo32
                   0, 0 };
        v4i gz4 = { 0, 0, 0, 0 };
        v8i gz8 = { 0, 0, 0, 0, 0, 0, 0, 0 };
        __builtin_amdgcn_tensor_load_to_lds(g0, g1, gz4, gz4, gz8, 0);
        __builtin_amdgcn_s_wait_tensorcnt(0);
    }
#else
    // Fallback: cooperative 128-bit copy, 16 tokens x 768 bf16 = 24KB
    {
        const uint4* src = (const uint4*)(Abf + (size_t)tok0 * D_MODEL);
        uint4*       dst = (uint4*)Ap;
        for (int i = tid; i < (16 * D_MODEL * 2) / 16; i += 256) dst[i] = src[i];
    }
#endif
    __syncthreads();

    // Register-resident running top-64 for 2 owned tokens (2 slots per lane)
    float tval[2][2]; int tidx[2][2];
    float minv[2];    int minid[2];
#pragma unroll
    for (int t = 0; t < 2; ++t) {
        tval[t][0] = -3.4e38f; tval[t][1] = -3.4e38f;
        tidx[t][0] = 0;        tidx[t][1] = 0;
        wave_min2(tval[t][0], tval[t][1], lane, minv[t], minid[t]);
    }

    for (int tile = 0; tile < D_SAE / 128; ++tile) {
        const int gcol = tile * 128 + w * 16 + l15;
        // warm L2 one column-panel ahead (global_prefetch_b8)
        if (tile + 1 < D_SAE / 128)
            __builtin_prefetch(Wt + (size_t)(gcol + 128) * D_MODEL, 0, 1);
        v8f acc = {};
        const bf16* arow = Ap + l15 * D_MODEL + hi * 16;             // A: M=l15
        const bf16* bcol = Wt + (size_t)gcol * D_MODEL + hi * 16;    // B: N=gcol
#pragma unroll 4
        for (int k0 = 0; k0 < D_MODEL; k0 += 32) {
            union { v16bf v; struct { v8bf lo, hi; } p; } a;
            a.p.lo = *(const v8bf*)(arow + k0);          // K=(0..7)+hi*16
            a.p.hi = *(const v8bf*)(arow + k0 + 8);      // K=(8..15)+hi*16
            v16bf b = *(const v16bf*)(bcol + k0);        // K contiguous 16
            acc = __builtin_amdgcn_wmma_f32_16x16x32_bf16(
                false, a.v, false, b, (short)0, acc, false, false);
        }
        // epilogue: + b_enc, stage candidates to LDS
        const float be = b_enc[gcol];
        float* st = stage + (hi * 8) * 128 + w * 16 + l15;
#pragma unroll
        for (int r = 0; r < 8; ++r) st[r * 128] = acc[r] + be;
        __syncthreads();

        // top-k update: wave w owns token rows 2w, 2w+1 of the stage
#pragma unroll
        for (int t = 0; t < 2; ++t) {
            const int row = 2 * w + t;
            const float4 c4 = *(const float4*)(stage + row * 128 + lane * 4);
            const int cb = tile * 128 + lane * 4;
            float cv[4] = {c4.x, c4.y, c4.z, c4.w};
#pragma unroll
            for (int j = 0; j < 4; ++j) {
                unsigned mask = (unsigned)__ballot(cv[j] > minv[t]);
                while (mask) {
                    const int   src = __ffs(mask) - 1;
                    const float bv  = __shfl(cv[j], src, 32);
                    const int   bc  = __shfl(cb + j, src, 32);
                    if (lane == (minid[t] >> 1)) {       // replace current min
                        tval[t][minid[t] & 1] = bv;
                        tidx[t][minid[t] & 1] = bc;
                    }
                    wave_min2(tval[t][0], tval[t][1], lane, minv[t], minid[t]);
                    mask &= ~(1u << src);
                    mask &= (unsigned)__ballot(cv[j] > minv[t]);  // prune
                }
            }
        }
        __syncthreads();
    }

    // emit top-64 (value, feature index) per owned token
#pragma unroll
    for (int t = 0; t < 2; ++t) {
        const int token = tok0 + 2 * w + t;
        topv[token * TOPK + lane * 2 + 0] = tval[t][0];
        topi[token * TOPK + lane * 2 + 0] = tidx[t][0];
        topv[token * TOPK + lane * 2 + 1] = tval[t][1];
        topi[token * TOPK + lane * 2 + 1] = tidx[t][1];
    }
}

// ---------------------------------------------------------------------------
// Kernel 3: sparse decode. recon[n] = b_dec + sum_k relu(v_k) * W_dec[idx_k].
// W_dec (75MB) is fully L2-resident -> L2-bound float4 gather-FMA.
// ---------------------------------------------------------------------------
__global__ void decode(const float* __restrict__ topv, const int* __restrict__ topi,
                       const float* __restrict__ W_dec, const float* __restrict__ b_dec,
                       float* __restrict__ out) {
    __shared__ float rv[TOPK];
    __shared__ int   ri[TOPK];
    const int token = blockIdx.x;
    const int tid   = threadIdx.x;   // 0..191, 4 floats each = 768
    if (tid < TOPK) {
        rv[tid] = fmaxf(topv[token * TOPK + tid], 0.0f);   // relu(top values)
        ri[tid] = topi[token * TOPK + tid];
    }
    __syncthreads();
    float4 acc = *(const float4*)(b_dec + tid * 4);
#pragma unroll 4
    for (int k = 0; k < TOPK; ++k) {
        const float4 wr = *(const float4*)(W_dec + (size_t)ri[k] * D_MODEL + tid * 4);
        const float  s  = rv[k];
        acc.x += s * wr.x; acc.y += s * wr.y; acc.z += s * wr.z; acc.w += s * wr.w;
    }
    *(float4*)(out + (size_t)token * D_MODEL + tid * 4) = acc;
}

// ---------------------------------------------------------------------------
extern "C" void kernel_launch(void* const* d_in, const int* in_sizes, int n_in,
                              void* d_out, int out_size, void* d_ws, size_t ws_size,
                              hipStream_t stream) {
    const float* acts  = (const float*)d_in[0];
    const float* W_enc = (const float*)d_in[1];
    const float* W_dec = (const float*)d_in[2];
    const float* b_enc = (const float*)d_in[3];
    const float* b_dec = (const float*)d_in[4];
    float* out = (float*)d_out;

    // workspace layout (~54.5 MB total)
    char* ws   = (char*)d_ws;
    bf16*  Abf  = (bf16*)(ws);                      // 8192*768*2   = 12,582,912
    bf16*  Wt   = (bf16*)(ws + 12582912);           // 24576*768*2  = 37,748,736
    float* topv = (float*)(ws + 50331648);          // 8192*64*4    =  2,097,152
    int*   topi = (int*)  (ws + 52428800);          // 8192*64*4    =  2,097,152

    cvt_acts<<<(NTOK * D_MODEL + 255) / 256, 256, 0, stream>>>(acts, Abf);
    cvt_wenc<<<dim3(D_SAE / 32, D_MODEL / 32), dim3(32, 8), 0, stream>>>(W_enc, Wt);
    encode_topk<<<NTOK / 16, 256, 0, stream>>>(Abf, Wt, b_enc, topv, topi);
    decode<<<NTOK, 192, 0, stream>>>(topv, topi, W_dec, b_dec, out);
}
